// SparseMoE_57140244906728
// MI455X (gfx1250) — matrix operand order repaired and checked by reading, not compiled, not run on hardware
//
#include <hip/hip_runtime.h>
#include <hip/hip_bf16.h>
#include <math.h>

// ---------------- problem constants ----------------
#define B_DIM 4
#define S_DIM 2048
#define D_DIM 1024
#define E_DIM 8
#define TOPK  2
#define H_DIM 2730                 // expert hidden
#define HP    2752                 // H padded to multiple of 32 (32*86)
#define NTOK  (B_DIM * S_DIM)      // 8192
#define CAP   (NTOK * TOPK / E_DIM) // 2048 per-expert capacity

typedef __attribute__((ext_vector_type(16))) __bf16 v16bf;
typedef __attribute__((ext_vector_type(8)))  float  v8f;

#define WMMA_BF16(A, B, C) \
    __builtin_amdgcn_wmma_f32_16x16x32_bf16(false, (A), false, (B), (short)0, (C), false, false)

// ---------------- WMMA fragment loaders (wave32 layouts, 16x16x32 bf16) ----
// A: 16x32 row-major, lane L: M = L&15, half = L>>4.
//    v[0..7]  = K in [8*half,   8*half+7]
//    v[8..15] = K in [16+8*half, 16+8*half+7]
__device__ __forceinline__ v16bf load_a_frag(const __bf16* row, int kBase, int half) {
    union { v16bf v; uint4 u[2]; } f;
    f.u[0] = *(const uint4*)(row + kBase + 8 * half);
    f.u[1] = *(const uint4*)(row + kBase + 16 + 8 * half);
    return f.v;
}
// B: 32x16; stored K-contiguous per output column. lane L: N = L&15, half = L>>4.
//    v[0..7]  = K in [8*half,   8*half+7]
//    v[8..15] = K in [16+8*half, 16+8*half+7]
__device__ __forceinline__ v16bf load_b_frag(const __bf16* colRow, int kBase, int half) {
    union { v16bf v; uint4 u[2]; } f;
    f.u[0] = *(const uint4*)(colRow + kBase + 8 * half);
    f.u[1] = *(const uint4*)(colRow + kBase + 16 + 8 * half);
    return f.v;
}

__device__ __forceinline__ float silu_mul(float a, float b) {
    return (a / (1.0f + __expf(-a))) * b;
}

// ---------------- utility kernels ----------------
__global__ void zero_f32_kernel(float* p, size_t n) {
    size_t i  = (size_t)blockIdx.x * blockDim.x + threadIdx.x;
    size_t st = (size_t)gridDim.x * blockDim.x;
    for (; i < n; i += st) p[i] = 0.0f;
}

__global__ void cvt_bf16_kernel(const float* __restrict__ s, __bf16* __restrict__ d, size_t n) {
    size_t i  = (size_t)blockIdx.x * blockDim.x + threadIdx.x;
    size_t st = (size_t)gridDim.x * blockDim.x;
    for (; i < n; i += st) d[i] = (__bf16)s[i];
}

// dst[e][orow][ocol] = (orow<C && ocol<R) ? src[e][ocol][colOff+orow] : 0   (f32 -> bf16)
__global__ void transpose_cvt_kernel(const float* __restrict__ src, size_t srcExpStride,
                                     int srcStride, int colOff, int R, int C,
                                     __bf16* __restrict__ dst, size_t dstExpStride,
                                     int OR_, int OC_) {
    __shared__ float tile[32][33];
    const float* s = src + (size_t)blockIdx.z * srcExpStride;
    __bf16*      d = dst + (size_t)blockIdx.z * dstExpStride;
    int ci0 = blockIdx.x * 32;   // output-row / src-col block
    int rj0 = blockIdx.y * 32;   // output-col / src-row block
    int tx = threadIdx.x, ty = threadIdx.y; // 32 x 8
    for (int yy = ty; yy < 32; yy += 8) {
        int r = rj0 + yy, c = ci0 + tx;
        tile[yy][tx] = (r < R && c < C) ? s[(size_t)r * srcStride + colOff + c] : 0.0f;
    }
    __syncthreads();
    for (int yy = ty; yy < 32; yy += 8) {
        int orow = ci0 + yy, ocol = rj0 + tx;
        if (orow < OR_ && ocol < OC_)
            d[(size_t)orow * OC_ + ocol] = (__bf16)tile[tx][yy];
    }
}

// ---------------- router ----------------
__global__ void router_kernel(const float* __restrict__ x, const float* __restrict__ noise,
                              const float* __restrict__ w_route, const float* __restrict__ b_route,
                              const float* __restrict__ w_noise, const float* __restrict__ b_noise,
                              int* __restrict__ i0a, int* __restrict__ i1a,
                              float* __restrict__ g0a, float* __restrict__ g1a,
                              float* __restrict__ esum) {
    __shared__ float wr[D_DIM * E_DIM];
    __shared__ float wn[D_DIM * E_DIM];
    __shared__ float ls[E_DIM];
    for (int i = threadIdx.x; i < D_DIM * E_DIM; i += blockDim.x) {
        wr[i] = w_route[i];
        wn[i] = w_noise[i];
    }
    if (threadIdx.x < E_DIM) ls[threadIdx.x] = 0.0f;
    __syncthreads();

    int tok = blockIdx.x * blockDim.x + threadIdx.x;
    float ar[E_DIM], an[E_DIM];
#pragma unroll
    for (int e = 0; e < E_DIM; ++e) { ar[e] = 0.0f; an[e] = 0.0f; }
    const float* xr = x + (size_t)tok * D_DIM;
    for (int d = 0; d < D_DIM; ++d) {
        float xv = xr[d];
#pragma unroll
        for (int e = 0; e < E_DIM; ++e) {
            ar[e] = fmaf(xv, wr[d * E_DIM + e], ar[e]);
            an[e] = fmaf(xv, wn[d * E_DIM + e], an[e]);
        }
    }
    float noisy[E_DIM];
#pragma unroll
    for (int e = 0; e < E_DIM; ++e) {
        float z  = an[e] + b_noise[e];
        float sp = fmaxf(z, 0.0f) + log1pf(__expf(-fabsf(z)));  // softplus
        noisy[e] = ar[e] + b_route[e] + noise[(size_t)tok * E_DIM + e] * sp;
    }
    // top-2 (first max wins ties, like lax.top_k)
    float v0 = -3.0e38f; int i0 = 0;
#pragma unroll
    for (int e = 0; e < E_DIM; ++e) if (noisy[e] > v0) { v0 = noisy[e]; i0 = e; }
    float v1 = -3.0e38f; int i1 = 0;
#pragma unroll
    for (int e = 0; e < E_DIM; ++e) if (e != i0 && noisy[e] > v1) { v1 = noisy[e]; i1 = e; }
    // softmax over the two values
    float e1 = __expf(v1 - v0);
    float p0 = 1.0f / (1.0f + e1);
    float p1 = e1 / (1.0f + e1);

    i0a[tok] = i0; i1a[tok] = i1; g0a[tok] = p0; g1a[tok] = p1;
    atomicAdd(&ls[i0], p0);
    atomicAdd(&ls[i1], p1);
    __syncthreads();
    if (threadIdx.x < E_DIM) atomicAdd(&esum[threadIdx.x], ls[threadIdx.x]);
}

__global__ void aux_loss_kernel(const float* __restrict__ esum, float* __restrict__ out_aux) {
    if (threadIdx.x == 0 && blockIdx.x == 0) {
        float l = 0.0f;
        for (int e = 0; e < E_DIM; ++e) {
            float m = esum[e] / (float)NTOK - 1.0f / (float)E_DIM;
            l += m * m;
        }
        *out_aux = l;
    }
}

// ---------------- per-expert capacity lists (one wave per expert) ----------
__global__ void build_lists_kernel(const int* __restrict__ i0a, const int* __restrict__ i1a,
                                   const float* __restrict__ g0a, const float* __restrict__ g1a,
                                   int* __restrict__ sel, float* __restrict__ gate,
                                   int* __restrict__ counts) {
    int e = blockIdx.x;
    int lane = threadIdx.x; // blockDim.x == 32
    int base = 0;
    for (int it = 0; it < NTOK / 32; ++it) {
        int t = it * 32 + lane;
        bool h0 = (i0a[t] == e);
        bool h1 = (i1a[t] == e);
        bool hit = h0 || h1;
        unsigned mask = (unsigned)__ballot(hit);
        int prefix = __popc(mask & ((1u << lane) - 1u));
        if (hit) {
            int pos = base + prefix;
            if (pos < CAP) {
                sel[e * CAP + pos]  = t;
                gate[e * CAP + pos] = h0 ? g0a[t] : g1a[t];
            }
        }
        base += __popc(mask);
    }
    int cnt = base < CAP ? base : CAP;
    if (lane == 0) counts[e] = cnt;
    for (int pos = cnt + lane; pos < CAP; pos += 32) {
        sel[e * CAP + pos]  = 0;
        gate[e * CAP + pos] = 0.0f;  // invalid slots contribute nothing
    }
}

// ---------------- GEMM up (x @ Wproj) fused with SwiGLU ----------------
// one wave computes a 32(tokens) x 32(hidden) tile for BOTH SwiGLU halves:
// 8 WMMAs per K-step on 6 fragment loads.  The second column group of each
// weight stream is addressed off the SAME base register via a constant
// 16*D element offset (folds into the 24-bit instruction offset).
__global__ void gemm_up_kernel(const __bf16* __restrict__ xb,
                               const __bf16* __restrict__ w1t, // [E][HP][D]
                               const __bf16* __restrict__ w2t, // [E][HP][D]
                               const int* __restrict__ sel,
                               __bf16* __restrict__ hidden) {  // [E][CAP][HP]
    const int e    = blockIdx.y;
    const int wave = threadIdx.x >> 5;
    const int lane = threadIdx.x & 31;
    const int Mt = CAP / 32, Nt = HP / 32;      // 64 x 86
    int tile = blockIdx.x * 8 + wave;
    if (tile >= Mt * Nt) return;
    int mt = tile / Nt, nt = tile % Nt;
    int m0 = mt * 32, n0 = nt * 32;
    int half = lane >> 4;
    int ml = lane & 15;

    int tok0 = sel[e * CAP + m0 + ml];
    int tok1 = sel[e * CAP + m0 + 16 + ml];
    const __bf16* a0row = xb + (size_t)tok0 * D_DIM;
    const __bf16* a1row = xb + (size_t)tok1 * D_DIM;
    const __bf16* b1 = w1t + ((size_t)e * HP + (n0 + ml)) * D_DIM;  // +16*D = col group 2
    const __bf16* b2 = w2t + ((size_t)e * HP + (n0 + ml)) * D_DIM;
    const int COLSTEP = 16 * D_DIM;  // constant -> immediate offset

    v8f c00 = {}, c01 = {}, c10 = {}, c11 = {};
    v8f d00 = {}, d01 = {}, d10 = {}, d11 = {};
    for (int k = 0; k < D_DIM; k += 32) {
        v16bf A0  = load_a_frag(a0row, k, half);
        v16bf A1  = load_a_frag(a1row, k, half);
        v16bf B1A = load_b_frag(b1, k, half);
        v16bf B1B = load_b_frag(b1 + COLSTEP, k, half);
        v16bf B2A = load_b_frag(b2, k, half);
        v16bf B2B = load_b_frag(b2 + COLSTEP, k, half);
        c00 = WMMA_BF16(A0, B1A, c00);
        c01 = WMMA_BF16(A0, B1B, c01);
        c10 = WMMA_BF16(A1, B1A, c10);
        c11 = WMMA_BF16(A1, B1B, c11);
        d00 = WMMA_BF16(A0, B2A, d00);
        d01 = WMMA_BF16(A0, B2B, d01);
        d10 = WMMA_BF16(A1, B2A, d10);
        d11 = WMMA_BF16(A1, B2B, d11);
    }
    __bf16* hb = hidden + (size_t)e * CAP * HP;
#pragma unroll
    for (int j = 0; j < 8; ++j) {
        int r0 = m0 + j + 8 * half;
        int r1 = r0 + 16;
        int ca = n0 + ml;
        int cb = ca + 16;
        hb[(size_t)r0 * HP + ca] = (__bf16)silu_mul(c00[j], d00[j]);
        hb[(size_t)r0 * HP + cb] = (__bf16)silu_mul(c01[j], d01[j]);
        hb[(size_t)r1 * HP + ca] = (__bf16)silu_mul(c10[j], d10[j]);
        hb[(size_t)r1 * HP + cb] = (__bf16)silu_mul(c11[j], d11[j]);
    }
}

// ---------------- GEMM down (hidden @ Wdown), gated scatter-add ----------
// one wave computes a 32(tokens) x 64(D) tile: 8 WMMAs per K-step on
// 2 A + 4 B fragment loads.  All 4 B streams and the second A stream are
// constant offsets from ONE base register each (fold into imm offsets).
__global__ void gemm_down_kernel(const __bf16* __restrict__ hidden, // [E][CAP][HP]
                                 const __bf16* __restrict__ wdt,    // [E][D][HP]
                                 const int* __restrict__ sel,
                                 const float* __restrict__ gate,
                                 float* __restrict__ out) {         // [NTOK][D]
    const int e    = blockIdx.y;
    const int wave = threadIdx.x >> 5;
    const int lane = threadIdx.x & 31;
    const int Mt = CAP / 32, Nt = D_DIM / 64;   // 64 x 16
    int tile = blockIdx.x * 8 + wave;
    if (tile >= Mt * Nt) return;
    int mt = tile / Nt, nt = tile % Nt;
    int m0 = mt * 32, n0 = nt * 64;
    int half = lane >> 4;
    int ml = lane & 15;

    const __bf16* a0 = hidden + ((size_t)e * CAP + (m0 + ml)) * HP;        // +16*HP = row group 2
    const __bf16* b0 = wdt + ((size_t)e * D_DIM + (n0 + ml)) * HP;         // +j*16*HP = col group j
    const int ROWSTEP = 16 * HP;  // constant -> immediate offset

    v8f acc0[4] = {{}, {}, {}, {}};   // rows m0..m0+15
    v8f acc1[4] = {{}, {}, {}, {}};   // rows m0+16..m0+31
    for (int k = 0; k < HP; k += 32) {
        v16bf A0 = load_a_frag(a0, k, half);
        v16bf A1 = load_a_frag(a0 + ROWSTEP, k, half);
#pragma unroll
        for (int j2 = 0; j2 < 4; ++j2) {
            v16bf B = load_b_frag(b0 + j2 * ROWSTEP, k, half);
            acc0[j2] = WMMA_BF16(A0, B, acc0[j2]);
            acc1[j2] = WMMA_BF16(A1, B, acc1[j2]);
        }
    }
#pragma unroll
    for (int j = 0; j < 8; ++j) {
        int r0 = m0 + j + 8 * half;
        int r1 = r0 + 16;
        int t0 = sel[e * CAP + r0];
        int t1 = sel[e * CAP + r1];
        float g0 = gate[e * CAP + r0];
        float g1 = gate[e * CAP + r1];
#pragma unroll
        for (int j2 = 0; j2 < 4; ++j2) {
            int c = n0 + ml + 16 * j2;
            atomicAdd(&out[(size_t)t0 * D_DIM + c], acc0[j2][j] * g0);
            atomicAdd(&out[(size_t)t1 * D_DIM + c], acc1[j2][j] * g1);
        }
    }
}

// ---------------- host launch ----------------
extern "C" void kernel_launch(void* const* d_in, const int* in_sizes, int n_in,
                              void* d_out, int out_size, void* d_ws, size_t ws_size,
                              hipStream_t stream) {
    const float* x       = (const float*)d_in[0];
    const float* noise   = (const float*)d_in[1];
    const float* w_route = (const float*)d_in[2];
    const float* b_route = (const float*)d_in[3];
    const float* w_noise = (const float*)d_in[4];
    const float* b_noise = (const float*)d_in[5];
    const float* w_proj  = (const float*)d_in[6];  // [E][D][2H]
    const float* w_down  = (const float*)d_in[7];  // [E][H][D]
    float* out = (float*)d_out;                    // [NTOK*D] + 1 aux

    char* ws = (char*)d_ws;
    size_t off = 0;
    auto wsalloc = [&](size_t bytes) -> void* {
        void* p = ws + off;
        off += (bytes + 255) & ~(size_t)255;
        return p;
    };
    __bf16* xb     = (__bf16*)wsalloc((size_t)NTOK * D_DIM * 2);
    __bf16* w1t    = (__bf16*)wsalloc((size_t)E_DIM * HP * D_DIM * 2);
    __bf16* w2t    = (__bf16*)wsalloc((size_t)E_DIM * HP * D_DIM * 2);
    __bf16* wdt    = (__bf16*)wsalloc((size_t)E_DIM * D_DIM * HP * 2);
    __bf16* hidden = (__bf16*)wsalloc((size_t)E_DIM * CAP * HP * 2);
    int*    i0a    = (int*)  wsalloc((size_t)NTOK * 4);
    int*    i1a    = (int*)  wsalloc((size_t)NTOK * 4);
    float*  g0a    = (float*)wsalloc((size_t)NTOK * 4);
    float*  g1a    = (float*)wsalloc((size_t)NTOK * 4);
    int*    sel    = (int*)  wsalloc((size_t)E_DIM * CAP * 4);
    float*  gate   = (float*)wsalloc((size_t)E_DIM * CAP * 4);
    int*    counts = (int*)  wsalloc(E_DIM * 4);
    float*  esum   = (float*)wsalloc(E_DIM * 4);

    // 1) zero output rows + routing sums (harness poisons, replays don't re-zero)
    zero_f32_kernel<<<2048, 256, 0, stream>>>(out, (size_t)NTOK * D_DIM);
    zero_f32_kernel<<<1, 32, 0, stream>>>(esum, E_DIM);

    // 2) precision conversion / weight transposes (K-contiguous bf16, zero-padded)
    cvt_bf16_kernel<<<2048, 256, 0, stream>>>(x, xb, (size_t)NTOK * D_DIM);
    {
        dim3 blk(32, 8);
        // W1^T: [E][HP][D] from w_proj cols [0,H)
        dim3 g1((HP + 31) / 32, (D_DIM + 31) / 32, E_DIM);
        transpose_cvt_kernel<<<g1, blk, 0, stream>>>(w_proj, (size_t)D_DIM * 2 * H_DIM,
                                                     2 * H_DIM, 0, D_DIM, H_DIM,
                                                     w1t, (size_t)HP * D_DIM, HP, D_DIM);
        // W2^T: [E][HP][D] from w_proj cols [H,2H)
        transpose_cvt_kernel<<<g1, blk, 0, stream>>>(w_proj, (size_t)D_DIM * 2 * H_DIM,
                                                     2 * H_DIM, H_DIM, D_DIM, H_DIM,
                                                     w2t, (size_t)HP * D_DIM, HP, D_DIM);
        // Wd^T: [E][D][HP] from w_down [E][H][D]
        dim3 g2((D_DIM + 31) / 32, (HP + 31) / 32, E_DIM);
        transpose_cvt_kernel<<<g2, blk, 0, stream>>>(w_down, (size_t)H_DIM * D_DIM,
                                                     D_DIM, 0, H_DIM, D_DIM,
                                                     wdt, (size_t)D_DIM * HP, D_DIM, HP);
    }

    // 3) router + aux loss
    router_kernel<<<NTOK / 256, 256, 0, stream>>>(x, noise, w_route, b_route, w_noise, b_noise,
                                                  i0a, i1a, g0a, g1a, esum);
    aux_loss_kernel<<<1, 1, 0, stream>>>(esum, out + (size_t)NTOK * D_DIM);

    // 4) capacity-limited dispatch lists (token order)
    build_lists_kernel<<<E_DIM, 32, 0, stream>>>(i0a, i1a, g0a, g1a, sel, gate, counts);

    // 5) expert FFNs: all experts in flight at once (blockIdx.y = expert)
    {
        dim3 gUp((CAP / 32) * (HP / 32) / 8, E_DIM);      // 688 x 8
        gemm_up_kernel<<<gUp, 256, 0, stream>>>(xb, w1t, w2t, sel, hidden);
        dim3 gDn((CAP / 32) * (D_DIM / 64) / 8, E_DIM);   // 128 x 8
        gemm_down_kernel<<<gDn, 256, 0, stream>>>(hidden, wdt, sel, gate, out);
    }
}
